// ScannedRNN_64072322122462
// MI455X (gfx1250) — matrix operand order repaired and checked by reading, not compiled, not run on hardware
//
#include <hip/hip_runtime.h>
#include <hip/hip_bf16.h>

// ScannedRNN (LSTM with resets) for MI455X / gfx1250.
// T=256, B=256, D=1024, H=1024.
// Per-step fused kernel: gates = [x_t ; h] @ [WiT ; WhT]^T + b via
// v_wmma_f32_16x16x32_bf16 (fp32 accumulate), fused reset masking and LSTM
// pointwise update. Weights converted once per launch to bf16 and transposed
// to (4H, K); they stay resident in the 192 MB L2 across all 256 steps.
//
// v3b: weight tiles are moved into LDS by the Tensor Data Mover
// (tensor_load_to_lds, TENSORcnt) instead of per-thread global->VGPR->LDS
// staging: one TDM descriptor per 32-K chunk DMAs a 3D tile
// (32 k x 128 n @ stride 1024 x 4 gates @ stride 1M elements) with D# padding
// (4 DWORDs after every 16) reproducing the bank-conflict-free stride-40
// LDS layout. Prefetched one chunk ahead into ping-pong LDS so the DMA
// overlaps WMMA compute. Activations (need cvt/masking) stay on the VALU path.
// This toolchain's builtin takes the 6-arg (clang-23) form.

typedef __attribute__((ext_vector_type(16))) __bf16 v16bf;
typedef __attribute__((ext_vector_type(8)))  float  v8f;
typedef __attribute__((ext_vector_type(4)))  unsigned int u32x4;
typedef __attribute__((ext_vector_type(8)))  int          i32x8;
typedef __attribute__((ext_vector_type(4)))  int          i32x4;

#define T_STEPS 256
#define B_SZ    256
#define D_SZ    1024
#define H_SZ    1024

// ---------------------------------------------------------------------------
// One-time (per launch) weight transpose + fp32->bf16: src (K,N) -> dst (N,K)
// ---------------------------------------------------------------------------
__global__ void xpose_bf16_kernel(const float* __restrict__ src,
                                  __bf16* __restrict__ dst,
                                  int K, int N) {
  int idx = blockIdx.x * blockDim.x + threadIdx.x;   // over N*K
  int n = idx / K;
  int k = idx - n * K;
  dst[idx] = (__bf16)src[(size_t)k * N + n];
}

// ---------------------------------------------------------------------------
// State init: c_ws = carry_c (fp32), h0 = bf16(carry_h)
// ---------------------------------------------------------------------------
__global__ void init_state_kernel(const float* __restrict__ carry_c,
                                  const float* __restrict__ carry_h,
                                  float* __restrict__ c,
                                  __bf16* __restrict__ h0, int n) {
  int i = blockIdx.x * blockDim.x + threadIdx.x;
  if (i < n) {
    c[i]  = carry_c[i];
    h0[i] = (__bf16)carry_h[i];
  }
}

// ---------------------------------------------------------------------------
// Final outputs: c_f = c_ws, h_f = ys[T-1]
// ---------------------------------------------------------------------------
__global__ void finalize_kernel(const float* __restrict__ c,
                                const float* __restrict__ ys_last,
                                float* __restrict__ cf,
                                float* __restrict__ hf, int n) {
  int i = blockIdx.x * blockDim.x + threadIdx.x;
  if (i < n) {
    cf[i] = c[i];
    hf[i] = ys_last[i];
  }
}

// ---------------------------------------------------------------------------
// Fused LSTM step.
// Block: 256 threads = 8 waves, each wave owns 16 hidden cols (x4 gates) and
// the full 32-row block tile (2 row-groups -> 8 v8f accumulators).
// Block tile: 32 batch rows x 128 hidden cols x 4 gates.
// 64 K-chunks of 32 (x@Wi then h@Wh), ping-pong LDS, TDM weight prefetch.
// ---------------------------------------------------------------------------
__launch_bounds__(256)
__global__ void lstm_step_kernel(const float* __restrict__ ins,          // (T,B,D)
                                 const unsigned char* __restrict__ resets,// (T,B) bool
                                 const __bf16* __restrict__ WiT,         // (4H, D)
                                 const __bf16* __restrict__ WhT,         // (4H, H)
                                 const float* __restrict__ bias,         // (4H)
                                 float* __restrict__ c,                  // (B,H)
                                 const __bf16* __restrict__ h_in,        // (B,H)
                                 __bf16* __restrict__ h_out,             // (B,H)
                                 float* __restrict__ ys,                 // (T,B,H)
                                 int t) {
  // Row stride 40 bf16 (80 B): 16-B aligned b128 reads, spread LDS banks.
  // TDM reproduces this layout with pad_enable (4 DWORDs after every 16).
  __shared__ __bf16 lA[2][32][40];    // activation tile: 32 rows x 32 k
  __shared__ __bf16 lB[2][512][40];   // weights: (4 gates x 128 n) x 32 k

  const int tid   = threadIdx.x;
  const int lane  = tid & 31;
  const int wave  = tid >> 5;                // 0..7 = column group
  const int brow0 = blockIdx.x * 32;         // batch row base
  const int n0    = blockIdx.y * 128;        // hidden col base (per gate)

  const float*         x_t = ins + (size_t)t * B_SZ * D_SZ;
  const unsigned char* r_t = resets + (size_t)t * B_SZ;

  // --- TDM weight-tile DMA: one 3D descriptor per 32-K chunk ---
  // dim0 = 32 bf16 (contiguous k), dim1 = 128 rows (stride 1024 elem),
  // dim2 = 4 gates (stride H*1024 = 1M elem). data_size = 2 bytes.
  // pad: after every 16 DWORDs (64 B row) insert 4 DWORDs -> 80 B LDS rows.
  auto issueB = [&](int ck, int buf) {
    const int ph = ck >> 5;
    const int kt = (ck & 31) * 32;
    const __bf16* Wt = ph ? WhT : WiT;
    const unsigned long long ga =
        (unsigned long long)(uintptr_t)(Wt + (size_t)n0 * 1024 + kt);
    const unsigned ldsa = (unsigned)(uintptr_t)(&lB[buf][0][0]);
    u32x4 g0;
    g0[0] = 1u;                                   // count=1, user descriptor
    g0[1] = ldsa;                                 // lds_addr
    g0[2] = (unsigned)ga;                         // global_addr[31:0]
    g0[3] = (unsigned)((ga >> 32) & 0x1FFFFFFu)   // global_addr[56:32]
            | (2u << 30);                         // type = 2 ("image")
    i32x8 g1;
    g1[0] = (1 << 16)                             // data_size = 2 bytes
            | (1 << 20)                           // pad_enable
            | (3 << 22)                           // pad_interval: 16 DWORDs
            | (3 << 25);                          // pad_amount: 4 DWORDs
    g1[1] = 32 << 16;                             // tensor_dim0 = 32 (lo16)
    g1[2] = 128 << 16;                            // tensor_dim1 = 128 (lo16)
    g1[3] = 32 << 16;                             // tile_dim0 = 32
    g1[4] = 128 | (4 << 16);                      // tile_dim1=128, tile_dim2=4
    g1[5] = 1024;                                 // tensor_dim0_stride lo32
    g1[6] = 0;                                    // strides hi bits
    g1[7] = (int)((1024ull * 1024ull) >> 16);     // tensor_dim1_stride[47:16]
    i32x4 g2;
    g2[0] = 4;                                    // tensor_dim2 = 4
    g2[1] = 0; g2[2] = 0; g2[3] = 0;
    i32x4 g3;
    g3[0] = 0; g3[1] = 0; g3[2] = 0; g3[3] = 0;
    i32x8 g4;                                     // extra descriptor words
    g4[0] = 0; g4[1] = 0; g4[2] = 0; g4[3] = 0;   // (6-arg clang-23 form,
    g4[4] = 0; g4[5] = 0; g4[6] = 0; g4[7] = 0;   //  zero-filled)
    __builtin_amdgcn_tensor_load_to_lds(g0, g1, g2, g3, g4, 0);
  };

  // --- activation tile fill (fp32->bf16 cvt / reset masking) ---
  auto fillA = [&](int ck, int buf) {
    const int ph = ck >> 5;
    const int kt = (ck & 31) * 32;
    if (ph == 0) {
      // x_t: 32 rows x 32 fp32 -> bf16. 256 thr x 4 floats.
      const int row = tid >> 3;
      const int seg = tid & 7;
      const float4 v = *reinterpret_cast<const float4*>(
          x_t + (size_t)(brow0 + row) * D_SZ + kt + seg * 4);
      __bf16* dst = &lA[buf][row][seg * 4];
      dst[0] = (__bf16)v.x; dst[1] = (__bf16)v.y;
      dst[2] = (__bf16)v.z; dst[3] = (__bf16)v.w;
    } else {
      // h: 32 rows x 32 bf16, reset-masked per batch row. 128 thr x 8 bf16.
      if (tid < 128) {
        const int row = tid >> 2;
        const int seg = tid & 3;
        uint4 v = *reinterpret_cast<const uint4*>(
            h_in + (size_t)(brow0 + row) * H_SZ + kt + seg * 8);
        if (r_t[brow0 + row]) v = make_uint4(0u, 0u, 0u, 0u);
        *reinterpret_cast<uint4*>(&lA[buf][row][seg * 8]) = v;
      }
    }
  };

  v8f acc[4][2];                             // [gate][row-group]
  #pragma unroll
  for (int g = 0; g < 4; ++g) {
    acc[g][0] = (v8f){};
    acc[g][1] = (v8f){};
  }

  // Prologue: chunk 0 into buffer 0.
  if (wave == 0) issueB(0, 0);
  fillA(0, 0);

  int pp = 0;
  for (int ck = 0; ck < 64; ++ck) {
    if (wave == 0) __builtin_amdgcn_s_wait_tensorcnt(0);  // TDM for chunk ck
    __syncthreads();                                      // buf pp ready
    if (ck + 1 < 64) {                                    // prefetch ck+1
      if (wave == 0) issueB(ck + 1, pp ^ 1);
      fillA(ck + 1, pp ^ 1);
    }

    // ---- WMMA compute from buffer pp ----
    // A fragment (16x32 bf16): lane<16 holds K {0..7,16..23}, lane>=16
    // holds K {8..15,24..31}; row = lane&15 within a 16-row group.
    const int arow   = lane & 15;
    const int kbaseA = (lane >= 16) ? 8 : 0;
    union Frag { uint4 u[2]; v16bf v; };
    Frag fa[2];
    #pragma unroll
    for (int rg = 0; rg < 2; ++rg) {
      fa[rg].u[0] =
          *reinterpret_cast<const uint4*>(&lA[pp][rg * 16 + arow][kbaseA]);
      fa[rg].u[1] = *reinterpret_cast<const uint4*>(
          &lA[pp][rg * 16 + arow][kbaseA + 16]);
    }
    // B fragment (32x16 bf16): lane<16 holds K {0..7,16..23}, lane>=16
    // holds K {8..15,24..31}; col = lane&15. lB is (n,k)-major -> two b128 reads.
    const int kbaseB = (lane >= 16) ? 8 : 0;
    #pragma unroll
    for (int g = 0; g < 4; ++g) {
      const __bf16* bp = &lB[pp][g * 128 + wave * 16 + (lane & 15)][kbaseB];
      Frag fb;
      fb.u[0] = *reinterpret_cast<const uint4*>(bp);
      fb.u[1] = *reinterpret_cast<const uint4*>(bp + 16);
      #pragma unroll
      for (int rg = 0; rg < 2; ++rg) {
        acc[g][rg] = __builtin_amdgcn_wmma_f32_16x16x32_bf16(
            false, fa[rg].v, false, fb.v, (short)0, acc[g][rg], false, false);
      }
    }
    pp ^= 1;
  }

  // ---- fused bias + LSTM pointwise update ----
  // C/D layout: v8f element vi -> row = vi + (lane>=16 ? 8 : 0), col = lane&15
  const int rowl = (lane >= 16) ? 8 : 0;
  const int n    = n0 + wave * 16 + (lane & 15);
  const float bi = bias[0 * H_SZ + n];
  const float bf = bias[1 * H_SZ + n];
  const float bg = bias[2 * H_SZ + n];
  const float bo = bias[3 * H_SZ + n];

  #pragma unroll
  for (int rg = 0; rg < 2; ++rg) {
    #pragma unroll
    for (int vi = 0; vi < 8; ++vi) {
      const int    row = brow0 + rg * 16 + rowl + vi;
      const size_t idx = (size_t)row * H_SZ + n;
      const float gi = acc[0][rg][vi] + bi;
      const float gf = acc[1][rg][vi] + bf;
      const float gg = acc[2][rg][vi] + bg;
      const float go = acc[3][rg][vi] + bo;
      const float cold = r_t[row] ? 0.0f : c[idx];
      const float si = 1.0f / (1.0f + __expf(-gi));
      const float sf = 1.0f / (1.0f + __expf(-gf));
      const float so = 1.0f / (1.0f + __expf(-go));
      const float cn = sf * cold + si * tanhf(gg);
      const float hn = so * tanhf(cn);
      c[idx]     = cn;
      h_out[idx] = (__bf16)hn;
      ys[(size_t)t * B_SZ * H_SZ + idx] = hn;
    }
  }
}

// ---------------------------------------------------------------------------
// Host launch
// ---------------------------------------------------------------------------
extern "C" void kernel_launch(void* const* d_in, const int* in_sizes, int n_in,
                              void* d_out, int out_size, void* d_ws, size_t ws_size,
                              hipStream_t stream) {
  const float*         carry_c = (const float*)d_in[0];
  const float*         carry_h = (const float*)d_in[1];
  const float*         ins     = (const float*)d_in[2];
  const unsigned char* resets  = (const unsigned char*)d_in[3]; // jax bool = 1 byte
  const float*         Wi      = (const float*)d_in[4];
  const float*         Wh      = (const float*)d_in[5];
  const float*         bias    = (const float*)d_in[6];

  // Workspace layout (18 MB total):
  //   [0,        8 MB)  WiT bf16 (4096 x 1024)
  //   [8 MB,    16 MB)  WhT bf16 (4096 x 1024)
  //   [16 MB,   17 MB)  c   fp32 (256 x 1024)
  //   [17 MB, +512 KB)  h buffer 0 (bf16)
  //   [...,   +512 KB)  h buffer 1 (bf16)
  char*   ws  = (char*)d_ws;
  __bf16* WiT = (__bf16*)(ws);
  __bf16* WhT = (__bf16*)(ws + (8ull << 20));
  float*  c   = (float*) (ws + (16ull << 20));
  __bf16* h0  = (__bf16*)(ws + (17ull << 20));
  __bf16* h1  = (__bf16*)(ws + (17ull << 20) + (512ull << 10));

  float* ys = (float*)d_out;                           // (T,B,H)
  float* cf = ys + (size_t)T_STEPS * B_SZ * H_SZ;      // (B,H)
  float* hf = cf + (size_t)B_SZ * H_SZ;                // (B,H)

  const int whel = 4 * H_SZ * D_SZ;                    // 4M elements each
  xpose_bf16_kernel<<<whel / 256, 256, 0, stream>>>(Wi, WiT, D_SZ, 4 * H_SZ);
  xpose_bf16_kernel<<<whel / 256, 256, 0, stream>>>(Wh, WhT, H_SZ, 4 * H_SZ);

  const int st = B_SZ * H_SZ;                          // 262144
  init_state_kernel<<<st / 256, 256, 0, stream>>>(carry_c, carry_h, c, h0, st);

  dim3 grid(B_SZ / 32, H_SZ / 128);                    // 8 x 8 = 64 blocks
  for (int t = 0; t < T_STEPS; ++t) {
    const __bf16* hin  = (t & 1) ? h1 : h0;
    __bf16*       hout = (t & 1) ? h0 : h1;
    lstm_step_kernel<<<grid, 256, 0, stream>>>(ins, resets, WiT, WhT, bias,
                                               c, hin, hout, ys, t);
  }

  finalize_kernel<<<st / 256, 256, 0, stream>>>(
      c, ys + (size_t)(T_STEPS - 1) * B_SZ * H_SZ, cf, hf, st);
}
